// AttentionLayer_18837726560562
// MI455X (gfx1250) — compile-verified, matrix operand order checked by reading, not run on hardware
//
#include <hip/hip_runtime.h>

// ---------------------------------------------------------------------------
// AttentionLayer for MI455X (gfx1250): all matmuls via v_wmma_f32_16x16x32_f16,
// B tiles staged in LDS with async global->LDS copies (ASYNCcnt), software-
// pipelined k-loops.
// ---------------------------------------------------------------------------

typedef _Float16 h16;
typedef __attribute__((ext_vector_type(16))) _Float16 v16h;
typedef __attribute__((ext_vector_type(8)))  _Float16 v8h;
typedef __attribute__((ext_vector_type(8)))  float    v8f;
typedef __attribute__((ext_vector_type(4)))  int      v4i;

#define DIMX   512
#define QKD    64
#define EXPD   1024       // EXPAND_DIM
#define OUTE   2176       // 2*QKD + 2*EXPD
#define NSEQ   2048
#define BATCH  8
#define NROWS  (BATCH * NSEQ)   // 16384

// ---- CDNA5 async copy: global -> LDS (16B per lane) -----------------------
// Builtin signature (from toolchain diagnostic): param0 = v4i* (addrspace 1
// global source), param1 = v4i* (addrspace 3 LDS dest), imm offset, imm cpol —
// same convention as __builtin_amdgcn_global_load_lds.

typedef __attribute__((address_space(1))) v4i* gv4p;   // global
typedef __attribute__((address_space(3))) v4i* lv4p;   // LDS (32-bit offset)

__device__ __forceinline__ void cp16_async(h16* lds_dst, const h16* gsrc) {
#if __has_builtin(__builtin_amdgcn_global_load_async_to_lds_b128)
  __builtin_amdgcn_global_load_async_to_lds_b128(
      (gv4p)(uintptr_t)gsrc,
      (lv4p)(uint32_t)(uintptr_t)lds_dst,   // low 32 bits of generic LDS addr
      0, 0);
#else
  *(v8h*)lds_dst = *(const v8h*)gsrc;          // sync fallback
#endif
}

template <int N>
__device__ __forceinline__ void wait_async() {
#if __has_builtin(__builtin_amdgcn_s_wait_asynccnt)
  __builtin_amdgcn_s_wait_asynccnt(N);
#endif
}

// ---- WMMA helpers ---------------------------------------------------------

__device__ __forceinline__ v8f wmma16(v16h a, v16h b, v8f c) {
  return __builtin_amdgcn_wmma_f32_16x16x32_f16(
      false, a, false, b, (short)0, c, false, false);
}

// A-fragment (16x32 row-major, ld) or B-fragment from Bt (N x K row-major).
// ISA 7.12.2: lane L -> row L&15; halves 0..7 = k[koff..koff+8),
// halves 8..15 = k[koff+16..koff+24), koff = (L>>4)*8.
__device__ __forceinline__ v16h load_frag(const h16* p, int ld, int lane) {
  const h16* q = p + (size_t)(lane & 15) * ld + ((lane >> 4) << 3);
  v8h lo = *(const v8h*)(q);
  v8h hi = *(const v8h*)(q + 16);
  v16h r;
#pragma unroll
  for (int i = 0; i < 8; ++i) { r[i] = lo[i]; r[i + 8] = hi[i]; }
  return r;
}

// D (16x16 f32): row = vgpr + 8*(lane>>4), col = lane&15.

// ---- Kernel 1: LayerNorm rows of x -> nx (f16) ----------------------------

__global__ void k_layernorm(const float* __restrict__ x, h16* __restrict__ nx) {
  const int row = blockIdx.x;
  const int t   = threadIdx.x;               // 256 threads, 2 elems each
  const float* xr = x + (size_t)row * DIMX;
  float v0 = xr[t], v1 = xr[t + 256];

  __shared__ float red[256];
  red[t] = v0 + v1;
  __syncthreads();
#pragma unroll
  for (int o = 128; o > 0; o >>= 1) {
    if (t < o) red[t] += red[t + o];
    __syncthreads();
  }
  const float mu = red[0] * (1.0f / DIMX);
  __syncthreads();

  const float d0 = v0 - mu, d1 = v1 - mu;
  red[t] = d0 * d0 + d1 * d1;
  __syncthreads();
#pragma unroll
  for (int o = 128; o > 0; o >>= 1) {
    if (t < o) red[t] += red[t + o];
    __syncthreads();
  }
  const float rstd = rsqrtf(red[0] * (1.0f / DIMX) + 1e-5f);

  h16* nr = nx + (size_t)row * DIMX;
  nr[t]       = (h16)(d0 * rstd);
  nr[t + 256] = (h16)(d1 * rstd);
}

// ---- Kernel 2a/2b: weight convert + transpose to Bt (N x K) f16 -----------

__global__ void k_convert_expand(const float* __restrict__ e, h16* __restrict__ eT) {
  int idx = blockIdx.x * 256 + threadIdx.x;
  if (idx < DIMX * OUTE) {
    int d = idx / OUTE, j = idx - d * OUTE;
    eT[(size_t)j * DIMX + d] = (h16)e[idx];
  }
}

__global__ void k_convert_proj(const float* __restrict__ p, h16* __restrict__ pT) {
  int idx = blockIdx.x * 256 + threadIdx.x;
  if (idx < EXPD * DIMX) {
    int e = idx / DIMX, o = idx - e * DIMX;
    pT[(size_t)o * EXPD + e] = (h16)p[idx];
  }
}

// ---- Kernel 3: q/k GEMM ---------------------------------------------------
// grid (NROWS/128, 2) x block 256 (8 waves). B tile (64 cols x 32 k) is
// async-staged in double-buffered LDS; A fragment register-prefetched.

__global__ void k_qk(const h16* __restrict__ nx, const h16* __restrict__ eT,
                     h16* __restrict__ qb, h16* __restrict__ kb) {
  __shared__ __align__(16) h16 sB[2][64 * 32];   // 4KB x2
  const int t    = threadIdx.x;
  const int wave = t >> 5, lane = t & 31;
  const int mbase = blockIdx.x * 128 + wave * 16;
  const int nc0   = blockIdx.y * 64;
  const h16* A = nx + (size_t)mbase * DIMX;

  // cooperative B copy: 64 rows x 4 chunks(16B) = 256 chunks, 1 per thread
  const int brow = t >> 2, bc = (t & 3) * 8;
  const h16* bsrc = eT + (size_t)(nc0 + brow) * DIMX + bc;
  h16* bdst[2] = { &sB[0][brow * 32 + bc], &sB[1][brow * 32 + bc] };

  const int KS = DIMX / 32;                      // 16 k-steps
  cp16_async(bdst[0], bsrc);                     // stage 0
  v8f acc[4] = {};
  v16h a_cur = load_frag(A, DIMX, lane);

  for (int s = 0; s < KS; ++s) {
    if (s + 1 < KS) { cp16_async(bdst[(s + 1) & 1], bsrc + (s + 1) * 32); wait_async<1>(); }
    else            { wait_async<0>(); }
    __syncthreads();                             // stage s visible to all waves
    v16h a_nxt;
    if (s + 1 < KS) a_nxt = load_frag(A + (s + 1) * 32, DIMX, lane);
    const h16* bt = &sB[s & 1][0];
#pragma unroll
    for (int nf = 0; nf < 4; ++nf) {
      v16h b = load_frag(bt + nf * 16 * 32, 32, lane);
      acc[nf] = wmma16(a_cur, b, acc[nf]);
    }
    a_cur = a_nxt;
    __syncthreads();                             // done reading before overwrite
  }

  const int coll = lane & 15, rb = (lane >> 4) * 8;
  const bool isq = (blockIdx.y == 0);
  h16* dst = isq ? qb : kb;
  const float sc = isq ? 0.125f : 1.0f;          // fold 1/sqrt(QKD) into q
#pragma unroll
  for (int nf = 0; nf < 4; ++nf)
#pragma unroll
    for (int r = 0; r < 8; ++r) {
      int row = mbase + rb + r;
      int col = nf * 16 + coll;
      dst[(size_t)row * QKD + col] = (h16)(acc[nf][r] * sc);
    }
}

// ---- Kernel 4: gate GEMM --------------------------------------------------
// Per wave 16x64 of BOTH linear (cols 128+j) and pre_gelu (cols 1152+j).
// LDS B tile: rows 0..63 = linear cols, rows 64..127 = gelu cols.
// grid (NROWS/128, 16) x block 256.

__global__ void k_gate(const h16* __restrict__ nx, const h16* __restrict__ eT,
                       h16* __restrict__ cat, h16* __restrict__ vT) {
  __shared__ __align__(16) h16 sB[2][128 * 32];  // 8KB x2
  const int t    = threadIdx.x;
  const int wave = t >> 5, lane = t & 31;
  const int mbase = blockIdx.x * 128 + wave * 16;
  const int j0    = blockIdx.y * 64;
  const h16* A = nx + (size_t)mbase * DIMX;

  // 128 rows x 4 chunks = 512 chunks, 2 per thread (chunk t and t+256)
  int cr0 = t >> 2,          cc0 = (t & 3) * 8;
  int cr1 = (t + 256) >> 2,  cc1 = cc0;
  int er0 = (cr0 < 64) ? (128 + j0 + cr0) : (1152 + j0 + cr0 - 64);
  int er1 = (cr1 < 64) ? (128 + j0 + cr1) : (1152 + j0 + cr1 - 64);
  const h16* bsrc0 = eT + (size_t)er0 * DIMX + cc0;
  const h16* bsrc1 = eT + (size_t)er1 * DIMX + cc1;
  h16* bd0[2] = { &sB[0][cr0 * 32 + cc0], &sB[1][cr0 * 32 + cc0] };
  h16* bd1[2] = { &sB[0][cr1 * 32 + cc1], &sB[1][cr1 * 32 + cc1] };

  const int KS = DIMX / 32;
  cp16_async(bd0[0], bsrc0);
  cp16_async(bd1[0], bsrc1);
  v8f accL[4] = {}, accG[4] = {};
  v16h a_cur = load_frag(A, DIMX, lane);

  for (int s = 0; s < KS; ++s) {
    if (s + 1 < KS) {
      cp16_async(bd0[(s + 1) & 1], bsrc0 + (s + 1) * 32);
      cp16_async(bd1[(s + 1) & 1], bsrc1 + (s + 1) * 32);
      wait_async<2>();
    } else wait_async<0>();
    __syncthreads();
    v16h a_nxt;
    if (s + 1 < KS) a_nxt = load_frag(A + (s + 1) * 32, DIMX, lane);
    const h16* bt = &sB[s & 1][0];
#pragma unroll
    for (int nf = 0; nf < 4; ++nf) {
      v16h bl = load_frag(bt + nf * 16 * 32, 32, lane);
      accL[nf] = wmma16(a_cur, bl, accL[nf]);
      v16h bg = load_frag(bt + (64 + nf * 16) * 32, 32, lane);
      accG[nf] = wmma16(a_cur, bg, accG[nf]);
    }
    a_cur = a_nxt;
    __syncthreads();
  }

  const int coll = lane & 15, rb = (lane >> 4) * 8;
#pragma unroll
  for (int nf = 0; nf < 4; ++nf)
#pragma unroll
    for (int r = 0; r < 8; ++r) {
      int row = mbase + rb + r;
      int j   = j0 + nf * 16 + coll;
      float lin = accL[nf][r], pg = accG[nf][r];
      float g   = 0.5f * pg * (1.0f + erff(pg * 0.70710678f));
      float val = lin * g;
      if (j < 512) {
        cat[(size_t)row * EXPD + j] = (h16)val;
      } else {
        int b = row >> 11, n = row & (NSEQ - 1);
        vT[((size_t)b * DIMX + (j - 512)) * NSEQ + n] = (h16)val;
      }
    }
}

// ---- Kernel 5: flash attention -------------------------------------------
// One block = one batch x one 64-row q tile. 16 waves (512 thr):
//   wave w: r = w&3 (q sub-rows), c = w>>2 (S k-cols / PV d-cols).
// q fragments hoisted out of the k-loop; softmax uses 8 threads/row.

__global__ void k_attn(const h16* __restrict__ qb, const h16* __restrict__ kb,
                       const h16* __restrict__ vT, const float* __restrict__ pbmp,
                       h16* __restrict__ cat) {
  const int q0    = blockIdx.x * 64;
  const int batch = blockIdx.y;
  const int wave  = threadIdx.x >> 5, lane = threadIdx.x & 31;
  const int r = wave & 3, c = wave >> 2;
  const int coll = lane & 15, rb = (lane >> 4) * 8;
  const float pbm = pbmp[0];

  __shared__ __align__(16) float sS[64 * 64];
  __shared__ __align__(16) h16   sP[64 * 64];
  __shared__ float sM[64], sL[64], sAlpha[64];

  if (threadIdx.x < 64) { sM[threadIdx.x] = -__builtin_inff(); sL[threadIdx.x] = 0.0f; }
  __syncthreads();

  v8f acc[8] = {};  // 16 rows x 128 cols (cols c*128 ..)

  const h16* qrow = qb + ((size_t)batch * NSEQ + q0 + r * 16) * QKD;
  v16h qa0 = load_frag(qrow,      QKD, lane);   // loop-invariant
  v16h qa1 = load_frag(qrow + 32, QKD, lane);

  for (int k0 = 0; k0 <= q0; k0 += 64) {
    // --- S tile: wave (r,c) -> rows r*16.., k-cols c*16.., K=64 ---
    const h16* krow = kb + ((size_t)batch * NSEQ + k0 + c * 16) * QKD;
    v8f s = {};
    s = wmma16(qa0, load_frag(krow,      QKD, lane), s);
    s = wmma16(qa1, load_frag(krow + 32, QKD, lane), s);
#pragma unroll
    for (int rr = 0; rr < 8; ++rr) {
      int qi = q0 + r * 16 + rb + rr;
      int ki = k0 + c * 16 + coll;
      float sv = s[rr];
      if (ki <= qi) {
        float rel = (float)(ki - qi);
        sv += 1.0f / (1.0f + expf(-(rel + pbm)));
      } else {
        sv = -__builtin_inff();
      }
      sS[(r * 16 + rb + rr) * 64 + (c * 16 + coll)] = sv;
    }
    __syncthreads();

    // --- online softmax: 8 threads per row, shfl reductions ---
    {
      int rowi = threadIdx.x >> 3;               // 0..63
      int sub  = threadIdx.x & 7;
      float m_old = sM[rowi];
      float mt = -__builtin_inff();
#pragma unroll
      for (int j = 0; j < 8; ++j) mt = fmaxf(mt, sS[rowi * 64 + sub * 8 + j]);
#pragma unroll
      for (int m = 1; m < 8; m <<= 1) mt = fmaxf(mt, __shfl_xor(mt, m, 32));
      float m_new = fmaxf(m_old, mt);
      float alpha = expf(m_old - m_new);          // 0 on first tile
      float sum = 0.0f;
#pragma unroll
      for (int j = 0; j < 8; ++j) {
        float p = expf(sS[rowi * 64 + sub * 8 + j] - m_new);
        sP[rowi * 64 + sub * 8 + j] = (h16)p;
        sum += p;
      }
#pragma unroll
      for (int m = 1; m < 8; m <<= 1) sum += __shfl_xor(sum, m, 32);
      if (sub == 0) {
        sM[rowi] = m_new;
        sL[rowi] = sL[rowi] * alpha + sum;
        sAlpha[rowi] = alpha;
      }
    }
    __syncthreads();

    // --- rescale accumulators, then P @ V ---
#pragma unroll
    for (int rr = 0; rr < 8; ++rr) {
      float a_ = sAlpha[r * 16 + rb + rr];
#pragma unroll
      for (int f = 0; f < 8; ++f) acc[f][rr] *= a_;
    }
#pragma unroll
    for (int kk = 0; kk < 64; kk += 32) {
      v16h pa = load_frag((const h16*)sP + (r * 16) * 64 + kk, 64, lane);
#pragma unroll
      for (int f = 0; f < 8; ++f) {
        int d = c * 128 + f * 16;
        v16h vb = load_frag(vT + ((size_t)batch * DIMX + d) * NSEQ + k0 + kk, NSEQ, lane);
        acc[f] = wmma16(pa, vb, acc[f]);
      }
    }
    __syncthreads();   // protect sS/sP for next tile
  }

  // --- finalize: attn = acc / l -> cat[:, 512:1024] ---
#pragma unroll
  for (int rr = 0; rr < 8; ++rr) {
    int lrow = r * 16 + rb + rr;
    float inv = 1.0f / sL[lrow];
    int grow = batch * NSEQ + q0 + lrow;
#pragma unroll
    for (int f = 0; f < 8; ++f) {
      int col = c * 128 + f * 16 + coll;
      cat[(size_t)grow * EXPD + 512 + col] = (h16)(acc[f][rr] * inv);
    }
  }
}

// ---- Kernel 6: out = x + cat @ project ------------------------------------
// grid (NROWS/128, 8) x block 256; LDS-staged B, pipelined, K=1024.

__global__ void k_project(const h16* __restrict__ cat, const h16* __restrict__ pT,
                          const float* __restrict__ x, float* __restrict__ out) {
  __shared__ __align__(16) h16 sB[2][64 * 32];
  const int t    = threadIdx.x;
  const int wave = t >> 5, lane = t & 31;
  const int mbase = blockIdx.x * 128 + wave * 16;
  const int n0    = blockIdx.y * 64;
  const h16* A = cat + (size_t)mbase * EXPD;

  const int brow = t >> 2, bc = (t & 3) * 8;
  const h16* bsrc = pT + (size_t)(n0 + brow) * EXPD + bc;
  h16* bdst[2] = { &sB[0][brow * 32 + bc], &sB[1][brow * 32 + bc] };

  const int KS = EXPD / 32;                      // 32 k-steps
  cp16_async(bdst[0], bsrc);
  v8f acc[4] = {};
  v16h a_cur = load_frag(A, EXPD, lane);

  for (int s = 0; s < KS; ++s) {
    if (s + 1 < KS) { cp16_async(bdst[(s + 1) & 1], bsrc + (s + 1) * 32); wait_async<1>(); }
    else            { wait_async<0>(); }
    __syncthreads();
    v16h a_nxt;
    if (s + 1 < KS) a_nxt = load_frag(A + (s + 1) * 32, EXPD, lane);
    const h16* bt = &sB[s & 1][0];
#pragma unroll
    for (int nf = 0; nf < 4; ++nf) {
      v16h b = load_frag(bt + nf * 16 * 32, 32, lane);
      acc[nf] = wmma16(a_cur, b, acc[nf]);
    }
    a_cur = a_nxt;
    __syncthreads();
  }

  const int coll = lane & 15, rb = (lane >> 4) * 8;
#pragma unroll
  for (int nf = 0; nf < 4; ++nf)
#pragma unroll
    for (int r = 0; r < 8; ++r) {
      size_t idx = (size_t)(mbase + rb + r) * DIMX + n0 + nf * 16 + coll;
      out[idx] = x[idx] + acc[nf][r];
    }
}

// ---- launcher -------------------------------------------------------------

extern "C" void kernel_launch(void* const* d_in, const int* in_sizes, int n_in,
                              void* d_out, int out_size, void* d_ws, size_t ws_size,
                              hipStream_t stream) {
  const float* x       = (const float*)d_in[0];
  const float* expand  = (const float*)d_in[1];
  const float* project = (const float*)d_in[2];
  const float* pbm     = (const float*)d_in[3];
  float* out = (float*)d_out;

  h16* ws = (h16*)d_ws;
  size_t off = 0;
  h16* nx  = ws + off; off += (size_t)NROWS * DIMX;
  h16* eT  = ws + off; off += (size_t)OUTE * DIMX;
  h16* pT  = ws + off; off += (size_t)DIMX * EXPD;
  h16* qb  = ws + off; off += (size_t)NROWS * QKD;
  h16* kb  = ws + off; off += (size_t)NROWS * QKD;
  h16* vT  = ws + off; off += (size_t)BATCH * DIMX * NSEQ;
  h16* cat = ws + off; off += (size_t)NROWS * EXPD;

  k_layernorm<<<NROWS, 256, 0, stream>>>(x, nx);
  k_convert_expand<<<(DIMX * OUTE + 255) / 256, 256, 0, stream>>>(expand, eT);
  k_convert_proj<<<(EXPD * DIMX + 255) / 256, 256, 0, stream>>>(project, pT);
  k_qk  <<<dim3(NROWS / 128, 2),  256, 0, stream>>>(nx, eT, qb, kb);
  k_gate<<<dim3(NROWS / 128, 16), 256, 0, stream>>>(nx, eT, cat, vT);
  k_attn<<<dim3(NSEQ / 64, BATCH), 512, 0, stream>>>(qb, kb, vT, pbm, cat);
  k_project<<<dim3(NROWS / 128, 8), 256, 0, stream>>>(cat, pT, x, out);
}
